// MyGRU_7267084664855
// MI455X (gfx1250) — compile-verified, matrix operand order checked by reading, not run on hardware
//
#include <hip/hip_runtime.h>
#include <hip/hip_bf16.h>

// Problem constants (match reference)
#define BV 256     // batch
#define SV 256     // seq len
#define HV 512     // hidden
#define OV 512     // output
#define K2H 1024   // 2*H

typedef __attribute__((ext_vector_type(16))) __bf16 v16bf;
typedef __attribute__((ext_vector_type(8)))  __bf16 bf8;
typedef __attribute__((ext_vector_type(8)))  float  v8f;

// ---------------------------------------------------------------------------
// Fragment loaders per CDNA5 WMMA VGPR layout (16-bit, 16x32 A / 32x16 B).
// Lane L in [0,16): row M=L, holds K=kb+0..7 (v0..3) and K=kb+16..23 (v4..7).
// Lane L in [16,32): row M=L-16, holds K=kb+8..15 and K=kb+24..31.
// B-matrix mirrors this with N (== weight output row) in place of M.
// Both chunks are 8 contiguous bf16 = 16 bytes -> global_load_b128.
// ---------------------------------------------------------------------------
__device__ __forceinline__ v16bf load_frag(const __bf16* __restrict__ rowPtr,
                                           int kb, int half) {
  bf8 lo = *(const bf8*)(rowPtr + kb + 8 * half);
  bf8 hi = *(const bf8*)(rowPtr + kb + 16 + 8 * half);
  return __builtin_shufflevector(lo, hi, 0, 1, 2, 3, 4, 5, 6, 7,
                                 8, 9, 10, 11, 12, 13, 14, 15);
}

__device__ __forceinline__ v8f wmma_bf16(v16bf a, v16bf b, v8f c) {
  return __builtin_amdgcn_wmma_f32_16x16x32_bf16(
      /*neg_a=*/false, a, /*neg_b=*/false, b,
      /*c_mod=*/(short)0, c, /*reuse_a=*/false, /*reuse_b=*/false);
}

__device__ __forceinline__ float sigmoidf_(float x) {
  return 1.0f / (1.0f + __expf(-x));
}

// ---------------------------------------------------------------------------
// One-time weight conversion f32 -> bf16 (stays resident in 192MB L2)
// ---------------------------------------------------------------------------
__global__ __launch_bounds__(256) void k_prep_w(
    const float* __restrict__ Wg, const float* __restrict__ Wh,
    const float* __restrict__ Wo, __bf16* __restrict__ Wgb,
    __bf16* __restrict__ Whb, __bf16* __restrict__ Wob) {
  int id = blockIdx.x * 256 + threadIdx.x;
  const int nWg = K2H * K2H;            // 1048576
  const int nWh = HV * K2H;             // 524288
  const int nWo = OV * HV;              // 262144
  if (id < nWg) {
    Wgb[id] = (__bf16)Wg[id];
  } else if (id < nWg + nWh) {
    int j = id - nWg;
    Whb[j] = (__bf16)Wh[j];
  } else if (id < nWg + nWh + nWo) {
    int j = id - nWg - nWh;
    Wob[j] = (__bf16)Wo[j];
  }
}

// h = 0 (f32) and gates A-matrix h-half = 0 (bf16)  [h0 == 0]
__global__ __launch_bounds__(256) void k_init(float* __restrict__ h,
                                              __bf16* __restrict__ Ag) {
  int id = blockIdx.x * 256 + threadIdx.x;  // < B*H
  h[id] = 0.0f;
  int b = id >> 9, j = id & (HV - 1);
  Ag[b * K2H + HV + j] = (__bf16)0.0f;
}

// Per-step embedding gather -> bf16 into A matrices (gates + cand, x-half)
__global__ __launch_bounds__(256) void k_prep_x(
    const int* __restrict__ x, const float* __restrict__ emb,
    __bf16* __restrict__ Ag, __bf16* __restrict__ Ac, int t) {
  int id = blockIdx.x * 256 + threadIdx.x;  // < B*H
  int b = id >> 9, j = id & (HV - 1);
  int idx = x[b * SV + t];
  __bf16 v = (__bf16)emb[idx * HV + j];
  Ag[b * K2H + j] = v;
  Ac[b * K2H + j] = v;
}

// ---------------------------------------------------------------------------
// Gates GEMM: G = Ag[B,2H] x Wg^T  (+bg), epilogue sigmoid.
// cols < H  -> z buffer (f32)
// cols >= H -> r; write r*h (bf16) into cand A-matrix second half
// 256 waves: 16 row-tiles x 16 col-groups (4 tiles each) over [256 x 1024].
// ---------------------------------------------------------------------------
__global__ __launch_bounds__(256) void k_gates(
    const __bf16* __restrict__ Ag, const __bf16* __restrict__ Wgb,
    const float* __restrict__ bg, const float* __restrict__ h,
    float* __restrict__ zbuf, __bf16* __restrict__ Ac) {
  int wave = (blockIdx.x * 256 + threadIdx.x) >> 5;
  int lane = threadIdx.x & 31;
  int rowTile = wave >> 4;        // 0..15
  int colGroup = wave & 15;       // 0..15
  int mBase = rowTile * 16;
  int half = lane >> 4;
  int r = lane & 15;

  const __bf16* aRow = Ag + (mBase + r) * K2H;
  v8f acc[4] = {v8f{}, v8f{}, v8f{}, v8f{}};

  for (int kb = 0; kb < K2H; kb += 32) {
    v16bf a = load_frag(aRow, kb, half);
#pragma unroll
    for (int c = 0; c < 4; ++c) {
      int col = (colGroup * 4 + c) * 16 + r;   // weight output row
      v16bf b = load_frag(Wgb + col * K2H, kb, half);
      acc[c] = wmma_bf16(a, b, acc[c]);
    }
  }

#pragma unroll
  for (int c = 0; c < 4; ++c) {
    int col = (colGroup * 4 + c) * 16 + r;
    float bias = bg[col];
    if (col < HV) {
#pragma unroll
      for (int i = 0; i < 8; ++i) {
        int row = mBase + i + 8 * half;
        zbuf[row * HV + col] = sigmoidf_(acc[c][i] + bias);
      }
    } else {
#pragma unroll
      for (int i = 0; i < 8; ++i) {
        int row = mBase + i + 8 * half;
        float rr = sigmoidf_(acc[c][i] + bias);
        float rh = rr * h[row * HV + (col - HV)];
        Ac[row * K2H + col] = (__bf16)rh;
      }
    }
  }
}

// ---------------------------------------------------------------------------
// Candidate GEMM: C = Ac[B,2H] x Wh^T (+bh), epilogue tanh + state update:
//   h_new = (1-z)*h + z*cand  -> h (f32, in-place), hn_bf (bf16 for out GEMM),
//   and gates A-matrix h-half (bf16 for next step).
// 128 waves: 16 row-tiles x 8 col-groups over [256 x 512].
// ---------------------------------------------------------------------------
__global__ __launch_bounds__(256) void k_cand(
    const __bf16* __restrict__ Ac, const __bf16* __restrict__ Whb,
    const float* __restrict__ bh, const float* __restrict__ zbuf,
    float* __restrict__ h, __bf16* __restrict__ hnbf,
    __bf16* __restrict__ Ag) {
  int wave = (blockIdx.x * 256 + threadIdx.x) >> 5;
  int lane = threadIdx.x & 31;
  int rowTile = wave >> 3;       // 0..15
  int colGroup = wave & 7;       // 0..7
  int mBase = rowTile * 16;
  int half = lane >> 4;
  int r = lane & 15;

  const __bf16* aRow = Ac + (mBase + r) * K2H;
  v8f acc[4] = {v8f{}, v8f{}, v8f{}, v8f{}};

  for (int kb = 0; kb < K2H; kb += 32) {
    v16bf a = load_frag(aRow, kb, half);
#pragma unroll
    for (int c = 0; c < 4; ++c) {
      int col = (colGroup * 4 + c) * 16 + r;
      v16bf b = load_frag(Whb + col * K2H, kb, half);
      acc[c] = wmma_bf16(a, b, acc[c]);
    }
  }

#pragma unroll
  for (int c = 0; c < 4; ++c) {
    int col = (colGroup * 4 + c) * 16 + r;
    float bias = bh[col];
#pragma unroll
    for (int i = 0; i < 8; ++i) {
      int row = mBase + i + 8 * half;
      float cand = tanhf(acc[c][i] + bias);
      float zz = zbuf[row * HV + col];
      float hv = h[row * HV + col];
      float hn = (1.0f - zz) * hv + zz * cand;
      h[row * HV + col] = hn;
      hnbf[row * HV + col] = (__bf16)hn;
      Ag[row * K2H + HV + col] = (__bf16)hn;   // next step's gates input
    }
  }
}

// ---------------------------------------------------------------------------
// Output GEMM: Y_t = hn[B,H] x Wo^T (+bo) -> d_out ys slab at step t.
// 128 waves: 16 row-tiles x 8 col-groups over [256 x 512], K=512.
// ---------------------------------------------------------------------------
__global__ __launch_bounds__(256) void k_out(
    const __bf16* __restrict__ hnbf, const __bf16* __restrict__ Wob,
    const float* __restrict__ bo, float* __restrict__ y, int t) {
  int wave = (blockIdx.x * 256 + threadIdx.x) >> 5;
  int lane = threadIdx.x & 31;
  int rowTile = wave >> 3;
  int colGroup = wave & 7;
  int mBase = rowTile * 16;
  int half = lane >> 4;
  int r = lane & 15;

  const __bf16* aRow = hnbf + (mBase + r) * HV;
  v8f acc[4] = {v8f{}, v8f{}, v8f{}, v8f{}};

  for (int kb = 0; kb < HV; kb += 32) {
    v16bf a = load_frag(aRow, kb, half);
#pragma unroll
    for (int c = 0; c < 4; ++c) {
      int col = (colGroup * 4 + c) * 16 + r;
      v16bf b = load_frag(Wob + col * HV, kb, half);
      acc[c] = wmma_bf16(a, b, acc[c]);
    }
  }

#pragma unroll
  for (int c = 0; c < 4; ++c) {
    int col = (colGroup * 4 + c) * 16 + r;
    float bias = bo[col];
#pragma unroll
    for (int i = 0; i < 8; ++i) {
      int row = mBase + i + 8 * half;           // batch index
      y[(row * SV + t) * OV + col] = acc[c][i] + bias;
    }
  }
}

// Final h -> d_out[0 : B*H]
__global__ __launch_bounds__(256) void k_final(const float* __restrict__ h,
                                               float* __restrict__ out) {
  int id = blockIdx.x * 256 + threadIdx.x;
  out[id] = h[id];
}

// ---------------------------------------------------------------------------
extern "C" void kernel_launch(void* const* d_in, const int* in_sizes, int n_in,
                              void* d_out, int out_size, void* d_ws,
                              size_t ws_size, hipStream_t stream) {
  const int*   x   = (const int*)d_in[0];
  const float* emb = (const float*)d_in[1];
  const float* Wg  = (const float*)d_in[2];
  const float* bg  = (const float*)d_in[3];
  const float* Wh  = (const float*)d_in[4];
  const float* bh  = (const float*)d_in[5];
  const float* Wo  = (const float*)d_in[6];
  const float* bo  = (const float*)d_in[7];
  float* out = (float*)d_out;

  // Workspace layout (bytes), all 256B-aligned
  char* ws = (char*)d_ws;
  __bf16* Wgb  = (__bf16*)(ws + 0);                 // 2 MB
  __bf16* Whb  = (__bf16*)(ws + 2097152);           // 1 MB
  __bf16* Wob  = (__bf16*)(ws + 3145728);           // 0.5 MB
  __bf16* Ag   = (__bf16*)(ws + 3670016);           // [B,2H] bf16, gates A
  __bf16* Ac   = (__bf16*)(ws + 4194304);           // [B,2H] bf16, cand A
  float*  zbuf = (float*)(ws + 4718592);            // [B,H] f32
  float*  h    = (float*)(ws + 5242880);            // [B,H] f32 state
  __bf16* hnbf = (__bf16*)(ws + 5767168);           // [B,H] bf16

  // One-time per call: weights -> bf16, h=0, gates A h-half = 0
  const int nW = K2H * K2H + HV * K2H + OV * HV;    // 1,835,008
  k_prep_w<<<(nW + 255) / 256, 256, 0, stream>>>(Wg, Wh, Wo, Wgb, Whb, Wob);
  k_init<<<(BV * HV) / 256, 256, 0, stream>>>(h, Ag);

  for (int t = 0; t < SV; ++t) {
    k_prep_x<<<(BV * HV) / 256, 256, 0, stream>>>(x, emb, Ag, Ac, t);
    k_gates<<<32, 256, 0, stream>>>(Ag, Wgb, bg, h, zbuf, Ac);
    k_cand<<<16, 256, 0, stream>>>(Ac, Whb, bh, zbuf, h, hnbf, Ag);
    k_out<<<16, 256, 0, stream>>>(hnbf, Wob, bo, out + BV * HV, t);
  }

  k_final<<<(BV * HV) / 256, 256, 0, stream>>>(h, out);
}
